// EncoderLayer_30855045055054
// MI455X (gfx1250) — 16-bit and 8-bit matrix operand order repaired and checked by reading, compiled, not run on hardware
//
#include <hip/hip_runtime.h>
#include <cmath>

typedef __attribute__((ext_vector_type(16))) _Float16 v16h;
typedef __attribute__((ext_vector_type(8)))  float    v8f;
typedef __attribute__((ext_vector_type(8)))  int      v8i;
typedef __attribute__((ext_vector_type(4)))  float    v4f;
typedef __attribute__((ext_vector_type(4)))  int      v4i;
typedef __attribute__((ext_vector_type(2)))  int      v2i;

#define TOKENS 2048
#define LSEQ   1024
#define DMODEL 1024
#define DINNER 2048
#define DSTATE 16
#define DTRANK 64
#define DFF    4096

union FragH { v16h v; v4f q[2]; };
union FragI { v8i  v; v2i q[4]; v4i p[2]; };

__global__ __launch_bounds__(256)
void gemm_f16_kernel(const _Float16* __restrict__ A,
                     const _Float16* __restrict__ W,
                     float* __restrict__ C, int M, int N, int K) {
  __shared__ __align__(16) _Float16 sA[2][64][32];
  const int tid     = threadIdx.x;
  const int lane    = tid & 31;
  const int wave    = tid >> 5;
  const int half_id = lane >> 4;
  const int mrow    = lane & 15;
  const int m0      = blockIdx.y * 64;
  const int n0      = blockIdx.x * 128 + wave * 16;
  const int n       = n0 + mrow;
  const int nc      = (n < N) ? n : (N - 1);
  const _Float16* wrow = W + (size_t)nc * K;

  const int srow = tid >> 2;
  const int scol = (tid & 3) * 8;

  v8f acc0 = {}, acc1 = {}, acc2 = {}, acc3 = {};

  *(v4f*)&sA[0][srow][scol] =
      *(const v4f*)(A + (size_t)(m0 + srow) * K + scol);
  __syncthreads();

  int buf = 0;
  for (int k0 = 0; k0 < K; k0 += 32) {
    if (k0 + 32 < K) {
      *(v4f*)&sA[buf ^ 1][srow][scol] =
          *(const v4f*)(A + (size_t)(m0 + srow) * K + (k0 + 32) + scol);
      __builtin_prefetch(wrow + k0 + 32, 0, 3);
    }
    FragH b;
    b.q[0] = *(const v4f*)(wrow + k0 + half_id * 8);
    b.q[1] = *(const v4f*)(wrow + k0 + 16 + half_id * 8);
    FragH a0, a1, a2, a3;
    a0.q[0] = *(const v4f*)&sA[buf][ 0 + mrow][half_id * 8];
    a0.q[1] = *(const v4f*)&sA[buf][ 0 + mrow][16 + half_id * 8];
    a1.q[0] = *(const v4f*)&sA[buf][16 + mrow][half_id * 8];
    a1.q[1] = *(const v4f*)&sA[buf][16 + mrow][16 + half_id * 8];
    a2.q[0] = *(const v4f*)&sA[buf][32 + mrow][half_id * 8];
    a2.q[1] = *(const v4f*)&sA[buf][32 + mrow][16 + half_id * 8];
    a3.q[0] = *(const v4f*)&sA[buf][48 + mrow][half_id * 8];
    a3.q[1] = *(const v4f*)&sA[buf][48 + mrow][16 + half_id * 8];
    acc0 = __builtin_amdgcn_wmma_f32_16x16x32_f16(false, a0.v, false, b.v, (short)0, acc0, false, false);
    acc1 = __builtin_amdgcn_wmma_f32_16x16x32_f16(false, a1.v, false, b.v, (short)0, acc1, false, false);
    acc2 = __builtin_amdgcn_wmma_f32_16x16x32_f16(false, a2.v, false, b.v, (short)0, acc2, false, false);
    acc3 = __builtin_amdgcn_wmma_f32_16x16x32_f16(false, a3.v, false, b.v, (short)0, acc3, false, false);
    __syncthreads();
    buf ^= 1;
  }

  if (n < N) {
#pragma unroll
    for (int r = 0; r < 8; ++r) {
      int mr = half_id * 8 + r;
      C[(size_t)(m0 +  0 + mr) * N + n] = acc0[r];
      C[(size_t)(m0 + 16 + mr) * N + n] = acc1[r];
      C[(size_t)(m0 + 32 + mr) * N + n] = acc2[r];
      C[(size_t)(m0 + 48 + mr) * N + n] = acc3[r];
    }
  }
}

__global__ __launch_bounds__(256)
void gemm_iu8_kernel(const signed char* __restrict__ Aq,
                     const signed char* __restrict__ Wq,
                     const float* __restrict__ xscale,
                     const float* __restrict__ wsum, float winv,
                     float* __restrict__ C, int M, int N, int K) {
  __shared__ __align__(16) signed char sA[2][64][64];
  const int tid     = threadIdx.x;
  const int lane    = tid & 31;
  const int wave    = tid >> 5;
  const int half_id = lane >> 4;
  const int mrow    = lane & 15;
  const int m0      = blockIdx.y * 64;
  const int n0      = blockIdx.x * 128 + wave * 16;
  const int n       = n0 + mrow;
  const int nc      = (n < N) ? n : (N - 1);
  const signed char* wrow = Wq + (size_t)nc * K;

  const int srow = tid >> 2;
  const int scol = (tid & 3) * 16;

  v8i acc0 = {}, acc1 = {}, acc2 = {}, acc3 = {};

  *(v4i*)&sA[0][srow][scol] =
      *(const v4i*)(Aq + (size_t)(m0 + srow) * K + scol);
  __syncthreads();

  int buf = 0;
  for (int k0 = 0; k0 < K; k0 += 64) {
    if (k0 + 64 < K) {
      *(v4i*)&sA[buf ^ 1][srow][scol] =
          *(const v4i*)(Aq + (size_t)(m0 + srow) * K + (k0 + 64) + scol);
      __builtin_prefetch(wrow + k0 + 64, 0, 3);
    }
    FragI b;
    b.p[0] = *(const v4i*)(wrow + k0 + half_id * 16);
    b.p[1] = *(const v4i*)(wrow + k0 + 32 + half_id * 16);
    FragI a0, a1, a2, a3;
#pragma unroll
    for (int c = 0; c < 4; ++c) {
      a0.q[c] = *(const v2i*)&sA[buf][ 0 + mrow][(c >> 1) * 32 + half_id * 16 + (c & 1) * 8];
      a1.q[c] = *(const v2i*)&sA[buf][16 + mrow][(c >> 1) * 32 + half_id * 16 + (c & 1) * 8];
      a2.q[c] = *(const v2i*)&sA[buf][32 + mrow][(c >> 1) * 32 + half_id * 16 + (c & 1) * 8];
      a3.q[c] = *(const v2i*)&sA[buf][48 + mrow][(c >> 1) * 32 + half_id * 16 + (c & 1) * 8];
    }
    acc0 = __builtin_amdgcn_wmma_i32_16x16x64_iu8(true, a0.v, true, b.v, acc0, false, false);
    acc1 = __builtin_amdgcn_wmma_i32_16x16x64_iu8(true, a1.v, true, b.v, acc1, false, false);
    acc2 = __builtin_amdgcn_wmma_i32_16x16x64_iu8(true, a2.v, true, b.v, acc2, false, false);
    acc3 = __builtin_amdgcn_wmma_i32_16x16x64_iu8(true, a3.v, true, b.v, acc3, false, false);
    __syncthreads();
    buf ^= 1;
  }

  if (n < N) {
    float ws_eff = fmaxf(wsum[0] * winv, 1e-5f);
#pragma unroll
    for (int r = 0; r < 8; ++r) {
      int mr = half_id * 8 + r;
      int ma = m0 + mr, mb = m0 + 16 + mr, mc = m0 + 32 + mr, md = m0 + 48 + mr;
      C[(size_t)ma * N + n] = (float)acc0[r] * ws_eff * xscale[ma];
      C[(size_t)mb * N + n] = (float)acc1[r] * ws_eff * xscale[mb];
      C[(size_t)mc * N + n] = (float)acc2[r] * ws_eff * xscale[mc];
      C[(size_t)md * N + n] = (float)acc3[r] * ws_eff * xscale[md];
    }
  }
}

__global__ void f32_to_f16_kernel(const float* __restrict__ in,
                                  _Float16* __restrict__ out, int n) {
  int i = blockIdx.x * 256 + threadIdx.x;
  if (i < n) out[i] = (_Float16)in[i];
}

__global__ void conv_silu_kernel(const float* __restrict__ u_raw,
                                 const float* __restrict__ conv_w,
                                 const float* __restrict__ conv_b,
                                 float* __restrict__ u,
                                 _Float16* __restrict__ u16) {
  int idx = blockIdx.x * 256 + threadIdx.x;
  int t = idx / DINNER, d = idx - t * DINNER;
  int l = t & (LSEQ - 1);
  float s = conv_b[d];
#pragma unroll
  for (int j = 0; j < 4; ++j) {
    int ll = l - 3 + j;
    if (ll >= 0) s += conv_w[d * 4 + j] * u_raw[(size_t)(t - 3 + j) * (2 * DINNER) + d];
  }
  float v = s / (1.f + __expf(-s));
  u[idx] = v;
  u16[idx] = (_Float16)v;
}

__global__ void dt16_kernel(const float* __restrict__ xdbl,
                            _Float16* __restrict__ dt16) {
  int idx = blockIdx.x * 256 + threadIdx.x;
  int t = idx >> 6, k = idx & 63;
  dt16[idx] = (_Float16)xdbl[(size_t)t * 96 + k];
}

__global__ void softplus_kernel(float* __restrict__ delta,
                                const float* __restrict__ dtb) {
  int idx = blockIdx.x * 256 + threadIdx.x;
  int d = idx & (DINNER - 1);
  float v = delta[idx] + dtb[d];
  delta[idx] = (v > 20.f) ? v : log1pf(__expf(v));
}

__global__ __launch_bounds__(256)
void scan_kernel(const float* __restrict__ delta, const float* __restrict__ u,
                 const float* __restrict__ xdbl,  const float* __restrict__ A_log,
                 const float* __restrict__ Dp,    const float* __restrict__ u_raw,
                 _Float16* __restrict__ y16) {
  __shared__ float sB[DSTATE], sC[DSTATE];
  int b = blockIdx.x >> 3;
  int d = (blockIdx.x & 7) * 256 + threadIdx.x;
  float Ad[DSTATE], h[DSTATE];
#pragma unroll
  for (int n = 0; n < DSTATE; ++n) { Ad[n] = -__expf(A_log[d * DSTATE + n]); h[n] = 0.f; }
  float Dd = Dp[d];
  for (int l = 0; l < LSEQ; ++l) {
    int t = b * LSEQ + l;
    if (threadIdx.x < DSTATE)
      sB[threadIdx.x] = xdbl[(size_t)t * 96 + DTRANK + threadIdx.x];
    else if (threadIdx.x < 2 * DSTATE)
      sC[threadIdx.x - DSTATE] = xdbl[(size_t)t * 96 + DTRANK + DSTATE + (threadIdx.x - DSTATE)];
    __syncthreads();
    float dl = delta[(size_t)t * DINNER + d];
    float ul = u[(size_t)t * DINNER + d];
    float acc = 0.f;
#pragma unroll
    for (int n = 0; n < DSTATE; ++n) {
      h[n] = __expf(Ad[n] * dl) * h[n] + dl * sB[n] * ul;
      acc += h[n] * sC[n];
    }
    float rv = u_raw[(size_t)t * (2 * DINNER) + DINNER + d];
    float yv = (acc + ul * Dd) * (rv / (1.f + __expf(-rv)));
    y16[(size_t)t * DINNER + d] = (_Float16)yv;
    __syncthreads();
  }
}

__global__ __launch_bounds__(256)
void add_rmsnorm_kernel(const float* __restrict__ a, const float* __restrict__ b,
                        const float* __restrict__ w, float* __restrict__ out) {
  __shared__ float red[256];
  int t = blockIdx.x;
  size_t base = (size_t)t * DMODEL;
  float vv[4], s = 0.f;
#pragma unroll
  for (int i = 0; i < 4; ++i) {
    int c = threadIdx.x + i * 256;
    float v = a[base + c] + b[base + c];
    vv[i] = v; s += v * v;
  }
  red[threadIdx.x] = s;
  __syncthreads();
  for (int off = 128; off; off >>= 1) {
    if (threadIdx.x < off) red[threadIdx.x] += red[threadIdx.x + off];
    __syncthreads();
  }
  float rs = rsqrtf(red[0] * (1.f / DMODEL) + 1e-6f);
#pragma unroll
  for (int i = 0; i < 4; ++i) {
    int c = threadIdx.x + i * 256;
    out[base + c] = vv[i] * rs * w[c];
  }
}

__global__ __launch_bounds__(256)
void quant_act_kernel(const float* __restrict__ x, signed char* __restrict__ xq,
                      float* __restrict__ xs, int cols) {
  __shared__ float red[256];
  int t = blockIdx.x;
  size_t base = (size_t)t * cols;
  float m = 0.f;
  for (int c = threadIdx.x; c < cols; c += 256) m = fmaxf(m, fabsf(x[base + c]));
  red[threadIdx.x] = m;
  __syncthreads();
  for (int off = 128; off; off >>= 1) {
    if (threadIdx.x < off) red[threadIdx.x] = fmaxf(red[threadIdx.x], red[threadIdx.x + off]);
    __syncthreads();
  }
  float xm = fmaxf(red[0], 1e-5f);
  float inv = 127.f / xm;
  for (int c = threadIdx.x; c < cols; c += 256) {
    float q = rintf(x[base + c] * inv);
    q = fminf(fmaxf(q, -128.f), 127.f);
    xq[base + c] = (signed char)q;
  }
  if (threadIdx.x == 0) xs[t] = xm * (1.f / 127.f);
}

__global__ void zero_kernel(float* p, int n) {
  int i = blockIdx.x * 256 + threadIdx.x;
  if (i < n) p[i] = 0.f;
}

__global__ __launch_bounds__(256)
void absmean_kernel(const float* __restrict__ w, int n, float* __restrict__ out) {
  __shared__ float red[256];
  float s = 0.f;
  for (int i = blockIdx.x * 256 + threadIdx.x; i < n; i += gridDim.x * 256)
    s += fabsf(w[i]);
  red[threadIdx.x] = s;
  __syncthreads();
  for (int off = 128; off; off >>= 1) {
    if (threadIdx.x < off) red[threadIdx.x] += red[threadIdx.x + off];
    __syncthreads();
  }
  if (threadIdx.x == 0) atomicAdd(out, red[0]);
}

__global__ void quant_w_kernel(const float* __restrict__ w, signed char* __restrict__ wq,
                               int n, const float* __restrict__ sum, float inv) {
  int i = blockIdx.x * 256 + threadIdx.x;
  if (i >= n) return;
  float ws = fmaxf(sum[0] * inv, 1e-5f);
  float q = rintf(w[i] / ws);
  q = fminf(fmaxf(q, -1.f), 1.f);
  wq[i] = (signed char)q;
}

__global__ void sigmul_kernel(float* __restrict__ gate, const float* __restrict__ up, int n) {
  int i = blockIdx.x * 256 + threadIdx.x;
  if (i >= n) return;
  float g = gate[i];
  gate[i] = up[i] / (1.f + __expf(-g));
}

extern "C" void kernel_launch(void* const* d_in, const int* in_sizes, int n_in,
                              void* d_out, int out_size, void* d_ws, size_t ws_size,
                              hipStream_t stream) {
  (void)in_sizes; (void)n_in; (void)out_size; (void)ws_size;
  const float* x          = (const float*)d_in[0];
  const float* in_proj_w  = (const float*)d_in[1];
  const float* conv_w     = (const float*)d_in[2];
  const float* conv_b     = (const float*)d_in[3];
  const float* x_proj_w   = (const float*)d_in[4];
  const float* dt_proj_w  = (const float*)d_in[5];
  const float* dt_proj_b  = (const float*)d_in[6];
  const float* A_log      = (const float*)d_in[7];
  const float* Dp         = (const float*)d_in[8];
  const float* out_proj_w = (const float*)d_in[9];
  const float* norm1_w    = (const float*)d_in[10];
  const float* gate_w     = (const float*)d_in[11];
  const float* up_w       = (const float*)d_in[12];
  const float* down_w     = (const float*)d_in[13];
  const float* norm2_w    = (const float*)d_in[14];
  float* out = (float*)d_out;

  char* p = (char*)d_ws;
  auto alloc = [&](size_t bytes) -> char* {
    char* r = p; p += (bytes + 255) & ~(size_t)255; return r;
  };

  _Float16* x16  = (_Float16*)alloc((size_t)TOKENS * DMODEL * 2);
  _Float16* w16a = (_Float16*)alloc((size_t)2 * DINNER * DMODEL * 2);
  _Float16* w16b = (_Float16*)alloc((size_t)(DTRANK + 2 * DSTATE) * DINNER * 2);
  _Float16* w16c = (_Float16*)alloc((size_t)DINNER * DTRANK * 2);
  _Float16* w16d = (_Float16*)alloc((size_t)DMODEL * DINNER * 2);
  float* u_raw   = (float*)alloc((size_t)TOKENS * 2 * DINNER * 4);
  float* u       = (float*)alloc((size_t)TOKENS * DINNER * 4);
  _Float16* u16  = (_Float16*)alloc((size_t)TOKENS * DINNER * 2);
  float* xdbl    = (float*)alloc((size_t)TOKENS * 96 * 4);
  _Float16* dt16 = (_Float16*)alloc((size_t)TOKENS * DTRANK * 2);
  float* delta   = (float*)alloc((size_t)TOKENS * DINNER * 4);
  _Float16* y16  = (_Float16*)alloc((size_t)TOKENS * DINNER * 2);
  float* mout    = (float*)alloc((size_t)TOKENS * DMODEL * 4);
  float* h       = (float*)alloc((size_t)TOKENS * DMODEL * 4);
  signed char* hq = (signed char*)alloc((size_t)TOKENS * DMODEL);
  float* hs      = (float*)alloc((size_t)TOKENS * 4);
  signed char* wq_gate = (signed char*)alloc((size_t)DFF * DMODEL);
  signed char* wq_up   = (signed char*)alloc((size_t)DFF * DMODEL);
  signed char* wq_down = (signed char*)alloc((size_t)DMODEL * DFF);
  float* wsum    = (float*)alloc(3 * 4);
  float* gate    = (float*)alloc((size_t)TOKENS * DFF * 4);
  float* up      = (float*)alloc((size_t)TOKENS * DFF * 4);
  signed char* gq = (signed char*)alloc((size_t)TOKENS * DFF);
  float* gs      = (float*)alloc((size_t)TOKENS * 4);
  float* fbuf    = (float*)alloc((size_t)TOKENS * DMODEL * 4);

  dim3 blk(256);
  auto nb = [](int n) { return (n + 255) / 256; };

  f32_to_f16_kernel<<<nb(TOKENS * DMODEL), blk, 0, stream>>>(x, x16, TOKENS * DMODEL);
  f32_to_f16_kernel<<<nb(2 * DINNER * DMODEL), blk, 0, stream>>>(in_proj_w, w16a, 2 * DINNER * DMODEL);
  f32_to_f16_kernel<<<nb((DTRANK + 2 * DSTATE) * DINNER), blk, 0, stream>>>(x_proj_w, w16b, (DTRANK + 2 * DSTATE) * DINNER);
  f32_to_f16_kernel<<<nb(DINNER * DTRANK), blk, 0, stream>>>(dt_proj_w, w16c, DINNER * DTRANK);
  f32_to_f16_kernel<<<nb(DMODEL * DINNER), blk, 0, stream>>>(out_proj_w, w16d, DMODEL * DINNER);

  gemm_f16_kernel<<<dim3((2 * DINNER) / 128, TOKENS / 64), blk, 0, stream>>>(
      x16, w16a, u_raw, TOKENS, 2 * DINNER, DMODEL);
  conv_silu_kernel<<<nb(TOKENS * DINNER), blk, 0, stream>>>(u_raw, conv_w, conv_b, u, u16);
  gemm_f16_kernel<<<dim3(1, TOKENS / 64), blk, 0, stream>>>(
      u16, w16b, xdbl, TOKENS, DTRANK + 2 * DSTATE, DINNER);
  dt16_kernel<<<nb(TOKENS * DTRANK), blk, 0, stream>>>(xdbl, dt16);
  gemm_f16_kernel<<<dim3(DINNER / 128, TOKENS / 64), blk, 0, stream>>>(
      dt16, w16c, delta, TOKENS, DINNER, DTRANK);
  softplus_kernel<<<nb(TOKENS * DINNER), blk, 0, stream>>>(delta, dt_proj_b);
  scan_kernel<<<16, blk, 0, stream>>>(delta, u, xdbl, A_log, Dp, u_raw, y16);
  gemm_f16_kernel<<<dim3(DMODEL / 128, TOKENS / 64), blk, 0, stream>>>(
      y16, w16d, mout, TOKENS, DMODEL, DINNER);
  add_rmsnorm_kernel<<<TOKENS, blk, 0, stream>>>(x, mout, norm1_w, h);

  quant_act_kernel<<<TOKENS, blk, 0, stream>>>(h, hq, hs, DMODEL);
  zero_kernel<<<1, blk, 0, stream>>>(wsum, 3);
  absmean_kernel<<<1024, blk, 0, stream>>>(gate_w, DFF * DMODEL, wsum + 0);
  absmean_kernel<<<1024, blk, 0, stream>>>(up_w, DFF * DMODEL, wsum + 1);
  absmean_kernel<<<1024, blk, 0, stream>>>(down_w, DMODEL * DFF, wsum + 2);
  quant_w_kernel<<<nb(DFF * DMODEL), blk, 0, stream>>>(gate_w, wq_gate, DFF * DMODEL, wsum + 0, 1.f / (DFF * DMODEL));
  quant_w_kernel<<<nb(DFF * DMODEL), blk, 0, stream>>>(up_w, wq_up, DFF * DMODEL, wsum + 1, 1.f / (DFF * DMODEL));
  quant_w_kernel<<<nb(DMODEL * DFF), blk, 0, stream>>>(down_w, wq_down, DMODEL * DFF, wsum + 2, 1.f / (DMODEL * DFF));

  gemm_iu8_kernel<<<dim3(DFF / 128, TOKENS / 64), blk, 0, stream>>>(
      hq, wq_gate, hs, wsum + 0, 1.f / (DFF * DMODEL), gate, TOKENS, DFF, DMODEL);
  gemm_iu8_kernel<<<dim3(DFF / 128, TOKENS / 64), blk, 0, stream>>>(
      hq, wq_up, hs, wsum + 1, 1.f / (DFF * DMODEL), up, TOKENS, DFF, DMODEL);
  sigmul_kernel<<<nb(TOKENS * DFF), blk, 0, stream>>>(gate, up, TOKENS * DFF);
  quant_act_kernel<<<TOKENS, blk, 0, stream>>>(gate, gq, gs, DFF);
  gemm_iu8_kernel<<<dim3(DMODEL / 128, TOKENS / 64), blk, 0, stream>>>(
      gq, wq_down, gs, wsum + 2, 1.f / (DMODEL * DFF), fbuf, TOKENS, DMODEL, DFF);
  add_rmsnorm_kernel<<<TOKENS, blk, 0, stream>>>(h, fbuf, norm2_w, out);
}
